// EncoderBlock_65790309040258
// MI455X (gfx1250) — matrix operand order repaired and checked by reading, not compiled, not run on hardware
//
#include <hip/hip_runtime.h>
#include <hip/hip_bf16.h>
#include <stdint.h>

// ---------------------------------------------------------------------------
// Problem constants (fixed by setup_inputs)
// ---------------------------------------------------------------------------
#define BB   2
#define SS   2048
#define DD   1024
#define HH   16
#define DQ   64
#define FF   4096
#define TT   (BB*SS)      // 4096 tokens
#define BH   (BB*HH)      // 32 batched heads
#define NEGV (-1e9f)

typedef __attribute__((ext_vector_type(16))) __bf16 bf16x16;
typedef __attribute__((ext_vector_type(8)))  float  f32x8;

struct FragRaw { uint4 a, b; };

static __device__ __forceinline__ bf16x16 make_frag(uint4 x, uint4 y) {
  FragRaw f{x, y};
  return __builtin_bit_cast(bf16x16, f);
}

static __device__ __forceinline__ unsigned short f2bf(float f) {
  unsigned int u = __float_as_uint(f);
  unsigned int r = u + 0x7FFFu + ((u >> 16) & 1u);   // round-to-nearest-even
  return (unsigned short)(r >> 16);
}
static __device__ __forceinline__ float bf2f(unsigned short h) {
  return __uint_as_float(((unsigned int)h) << 16);
}

// ---------------------------------------------------------------------------
// Weight conversion: fp32 -> bf16 hi/lo split
// ---------------------------------------------------------------------------
__global__ __launch_bounds__(256) void convert_wqkv_kernel(
    const float* __restrict__ wq, const float* __restrict__ wk,
    const float* __restrict__ wv,
    unsigned short* __restrict__ oh, unsigned short* __restrict__ ol) {
  int idx = blockIdx.x * 256 + threadIdx.x;      // n*1024 + k, n in [0,3072)
  int n = idx >> 10, k = idx & 1023;
  const float* src = (n < 1024) ? wq : ((n < 2048) ? wk : wv);
  int nn = n & 1023;
  // wq[h][d][q] with h=nn>>6, q=nn&63, d=k
  float v = src[(((size_t)(nn >> 6)) * 1024 + k) * 64 + (nn & 63)];
  unsigned short hi = f2bf(v);
  oh[idx] = hi;
  ol[idx] = f2bf(v - bf2f(hi));
}

__global__ __launch_bounds__(256) void convert_direct_kernel(
    const float* __restrict__ src,
    unsigned short* __restrict__ oh, unsigned short* __restrict__ ol, int count) {
  int idx = blockIdx.x * 256 + threadIdx.x;
  if (idx >= count) return;
  float v = src[idx];
  unsigned short hi = f2bf(v);
  oh[idx] = hi;
  ol[idx] = f2bf(v - bf2f(hi));
}

// ---------------------------------------------------------------------------
// LayerNorm (torch-style: unbiased std, scalar alpha/bias) -> bf16 hi/lo
// One block per row of 1024.
// ---------------------------------------------------------------------------
__global__ __launch_bounds__(256) void ln_split_kernel(
    const float* __restrict__ x, const float* __restrict__ alpha,
    const float* __restrict__ beta,
    unsigned short* __restrict__ oh, unsigned short* __restrict__ ol) {
  int row = blockIdx.x;
  const float* xr = x + (size_t)row * DD;
  float vals[4];
  float s = 0.f, q = 0.f;
#pragma unroll
  for (int i = 0; i < 4; i++) {
    float v = xr[threadIdx.x + i * 256];
    vals[i] = v; s += v; q += v * v;
  }
#pragma unroll
  for (int off = 16; off; off >>= 1) {
    s += __shfl_xor(s, off);
    q += __shfl_xor(q, off);
  }
  __shared__ float ss[8], sq[8];
  int w = threadIdx.x >> 5, l = threadIdx.x & 31;
  if (l == 0) { ss[w] = s; sq[w] = q; }
  __syncthreads();
  if (threadIdx.x == 0) {
    float S = 0.f, Q = 0.f;
#pragma unroll
    for (int i = 0; i < 8; i++) { S += ss[i]; Q += sq[i]; }
    ss[0] = S; sq[0] = Q;
  }
  __syncthreads();
  s = ss[0]; q = sq[0];
  float mean = s * (1.0f / DD);
  float var  = (q - (float)DD * mean * mean) * (1.0f / (DD - 1));
  float denom = sqrtf(fmaxf(var, 0.f)) + 1e-6f;
  float sc = alpha[0] / denom;
  float bi = beta[0];
#pragma unroll
  for (int i = 0; i < 4; i++) {
    float y = sc * (vals[i] - mean) + bi;
    size_t idx = (size_t)row * DD + threadIdx.x + i * 256;
    unsigned short hi = f2bf(y);
    oh[idx] = hi;
    ol[idx] = f2bf(y - bf2f(hi));
  }
}

// ---------------------------------------------------------------------------
// Generic bf16x3 GEMM:  C[M,N] = A[M,K] * B[N,K]^T   (both bf16 hi/lo)
// Block: 256 threads = 8 waves; block tile 128x128; wave tile 32x64.
// LDA/LDB/KK are compile-time so all fragment loads become immediate-offset
// clause-batched b128 loads off a single per-wave base pointer.
// ---------------------------------------------------------------------------
enum { M_QKV = 0, M_ATTN = 1, M_H = 2, M_ACC = 3 };

template <int MODE, int LDA, int LDB, int KK>
__global__ __launch_bounds__(256) void gemm_bf16x3_kernel(
    const unsigned short* __restrict__ Ah, const unsigned short* __restrict__ Al,
    const unsigned short* __restrict__ Bh, const unsigned short* __restrict__ Bl,
    float* __restrict__ outF,            // QKV: V fp32 | ATTN: x1 | ACC: d_out
    float* __restrict__ outF2,           // ATTN: d_out
    const float* __restrict__ resid,     // ATTN: x
    const float* __restrict__ bias,      // ATTN: b2 | H: b1 slice
    unsigned short* __restrict__ Oh,     // QKV: Qh | H: h_hi
    unsigned short* __restrict__ Ol) {   // QKV: Kh | H: h_lo
  int wave = threadIdx.x >> 5;
  int lane = threadIdx.x & 31;
  int lr = lane & 15, lh = lane >> 4;
  int wm = wave >> 1, wn = wave & 1;
  int m0 = blockIdx.y * 128 + wm * 32;
  int n0 = blockIdx.x * 128 + wn * 64;

  // Per-wave/lane base pointers; all further offsets are compile-time.
  const unsigned short* pah = Ah + (size_t)(m0 + lr) * LDA + lh * 8;
  const unsigned short* pal = Al + (size_t)(m0 + lr) * LDA + lh * 8;
  const unsigned short* pbh = Bh + (size_t)(n0 + lr) * LDB + lh * 8;
  const unsigned short* pbl = Bl + (size_t)(n0 + lr) * LDB + lh * 8;

  f32x8 acc[2][4] = {};

#pragma unroll 2
  for (int k0 = 0; k0 < KK; k0 += 32) {
    // A fragments (16x32 bf16): lanes 0-15 hold K{0-7,16-23}, 16-31 hold K{8-15,24-31}
    bf16x16 afh[2], afl[2];
#pragma unroll
    for (int mi = 0; mi < 2; mi++) {
      size_t off = (size_t)mi * 16 * LDA + k0;
      afh[mi] = make_frag(*(const uint4*)(pah + off), *(const uint4*)(pah + off + 16));
      afl[mi] = make_frag(*(const uint4*)(pal + off), *(const uint4*)(pal + off + 16));
    }
#pragma unroll
    for (int ni = 0; ni < 4; ni++) {
      size_t off = (size_t)ni * 16 * LDB + k0;
      bf16x16 bfh = make_frag(*(const uint4*)(pbh + off), *(const uint4*)(pbh + off + 16));
      bf16x16 bfl = make_frag(*(const uint4*)(pbl + off), *(const uint4*)(pbl + off + 16));
#pragma unroll
      for (int mi = 0; mi < 2; mi++) {
        acc[mi][ni] = __builtin_amdgcn_wmma_f32_16x16x32_bf16(
            false, afh[mi], false, bfh, (short)0, acc[mi][ni], false, false);
        acc[mi][ni] = __builtin_amdgcn_wmma_f32_16x16x32_bf16(
            false, afh[mi], false, bfl, (short)0, acc[mi][ni], false, false);
        acc[mi][ni] = __builtin_amdgcn_wmma_f32_16x16x32_bf16(
            false, afl[mi], false, bfh, (short)0, acc[mi][ni], false, false);
      }
    }
  }

  // Epilogue. C layout: lane holds column n0+ni*16+lr, rows m0+mi*16+lh*8+v.
#pragma unroll
  for (int mi = 0; mi < 2; mi++) {
#pragma unroll
    for (int ni = 0; ni < 4; ni++) {
      int col = n0 + ni * 16 + lr;
#pragma unroll
      for (int v = 0; v < 8; v++) {
        int row = m0 + mi * 16 + lh * 8 + v;
        float c = acc[mi][ni][v];
        if (MODE == M_QKV) {
          int b = row >> 11, s = row & (SS - 1);
          if (col < 1024) {
            Oh[(((size_t)(b * HH + (col >> 6))) * SS + s) * DQ + (col & 63)] = f2bf(c);
          } else if (col < 2048) {
            int cc = col - 1024;
            Ol[(((size_t)(b * HH + (cc >> 6))) * SS + s) * DQ + (cc & 63)] = f2bf(c);
          } else {
            outF[(size_t)row * DD + (col - 2048)] = c;
          }
        } else if (MODE == M_ATTN) {
          size_t idx = (size_t)row * DD + col;
          float t = c + resid[idx];
          outF[idx]  = t;                 // x1 (for LN2 + later)
          outF2[idx] = t + bias[col];     // d_out = x1 + b2 (FFN chunks add on)
        } else if (MODE == M_H) {
          size_t idx = (size_t)row * 1024 + col;
          float t = fmaxf(c + bias[col], 0.f);
          unsigned short hi = f2bf(t);
          Oh[idx] = hi;
          Ol[idx] = f2bf(t - bf2f(hi));
        } else {  // M_ACC
          size_t idx = (size_t)row * DD + col;
          outF[idx] += c;
        }
      }
    }
  }
}

// ---------------------------------------------------------------------------
// Fused attention-row statistics: online softmax (max, sumexp) per query row.
// Wave handles 16 query rows of one (b,h); sweeps t in 64-wide WMMA tiles.
// ---------------------------------------------------------------------------
__global__ __launch_bounds__(256) void attn_stats_kernel(
    const unsigned short* __restrict__ Qh, const unsigned short* __restrict__ Kh,
    const int* __restrict__ mask,
    float* __restrict__ rm, float* __restrict__ rl) {
  int bh = blockIdx.y;
  int b = bh >> 4;
  int wave = threadIdx.x >> 5, lane = threadIdx.x & 31;
  int lr = lane & 15, lh = lane >> 4;
  int rowBase = blockIdx.x * 128 + wave * 16;

  const unsigned short* Qb = Qh + ((size_t)bh * SS + rowBase) * DQ;
  const unsigned short* Kb = Kh + (size_t)bh * SS * DQ;

  // A fragments for this wave's 16 rows (K = dq = 64 -> two K-steps), reused all t.
  bf16x16 afr[2];
#pragma unroll
  for (int ks = 0; ks < 2; ks++) {
    const unsigned short* p = Qb + (size_t)lr * DQ + ks * 32 + lh * 8;
    afr[ks] = make_frag(*(const uint4*)p, *(const uint4*)(p + 16));
  }

  float m[8], l[8];
#pragma unroll
  for (int r = 0; r < 8; r++) { m[r] = -INFINITY; l[r] = 0.f; }

  for (int t0 = 0; t0 < SS; t0 += 64) {
    f32x8 acc[4] = {};
#pragma unroll
    for (int ti = 0; ti < 4; ti++) {
      int col = t0 + ti * 16 + lr;
#pragma unroll
      for (int ks = 0; ks < 2; ks++) {
        const unsigned short* p = Kb + (size_t)col * DQ + ks * 32 + lh * 8;
        bf16x16 bfr = make_frag(*(const uint4*)p, *(const uint4*)(p + 16));
        acc[ti] = __builtin_amdgcn_wmma_f32_16x16x32_bf16(
            false, afr[ks], false, bfr, (short)0, acc[ti], false, false);
      }
    }
#pragma unroll
    for (int ti = 0; ti < 4; ti++) {
      int col = t0 + ti * 16 + lr;            // all 8 values share this column
      bool mk = mask[b * SS + col] != 0;
#pragma unroll
      for (int r = 0; r < 8; r++) {
        float v = mk ? acc[ti][r] * 0.125f : NEGV;   // /sqrt(dq)=1/8
        float nm = fmaxf(m[r], v);
        l[r] = l[r] * __expf(m[r] - nm) + __expf(v - nm);
        m[r] = nm;
      }
    }
  }

  // Merge (m,l) across the 16 lanes sharing each row (xor<=8 stays in half).
#pragma unroll
  for (int off = 1; off < 16; off <<= 1) {
#pragma unroll
    for (int r = 0; r < 8; r++) {
      float om = __shfl_xor(m[r], off);
      float ol = __shfl_xor(l[r], off);
      float nm = fmaxf(m[r], om);
      l[r] = l[r] * __expf(m[r] - nm) + ol * __expf(om - nm);
      m[r] = nm;
    }
  }
  if (lr == 0) {
#pragma unroll
    for (int r = 0; r < 8; r++) {
      size_t idx = (size_t)bh * SS + rowBase + lh * 8 + r;
      rm[idx] = m[r];
      rl[idx] = l[r];
    }
  }
}

// ---------------------------------------------------------------------------
// Diagonal softmax weight * V -> attn (bf16 hi/lo). One wave per (b,h,s).
// Uses the SAME bf16 Q/K values as the stats kernel for consistency.
// ---------------------------------------------------------------------------
__global__ __launch_bounds__(256) void diag_v_kernel(
    const unsigned short* __restrict__ Qh, const unsigned short* __restrict__ Kh,
    const float* __restrict__ V, const int* __restrict__ mask,
    const float* __restrict__ rm, const float* __restrict__ rl,
    unsigned short* __restrict__ ah, unsigned short* __restrict__ al) {
  int widx = blockIdx.x * 8 + (threadIdx.x >> 5);   // (bh)*S + s
  int lane = threadIdx.x & 31;
  int bh = widx >> 11, s = widx & (SS - 1);
  int b = bh >> 4, h = bh & 15;
  const unsigned short* q = Qh + (size_t)widx * DQ;
  const unsigned short* k = Kh + (size_t)widx * DQ;
  float p = bf2f(q[lane]) * bf2f(k[lane]) + bf2f(q[lane + 32]) * bf2f(k[lane + 32]);
#pragma unroll
  for (int off = 16; off; off >>= 1) p += __shfl_xor(p, off);
  float ds = (mask[b * SS + s] != 0) ? p * 0.125f : NEGV;
  float dw = __expf(ds - rm[widx]) / rl[widx];
  size_t base = ((size_t)(b * SS + s)) * DD + h * DQ;
#pragma unroll
  for (int i = 0; i < 2; i++) {
    int qq = lane + i * 32;
    float o = dw * V[base + qq];
    unsigned short hi = f2bf(o);
    ah[base + qq] = hi;
    al[base + qq] = f2bf(o - bf2f(hi));
  }
}

// ---------------------------------------------------------------------------
// Launcher
// ---------------------------------------------------------------------------
extern "C" void kernel_launch(void* const* d_in, const int* in_sizes, int n_in,
                              void* d_out, int out_size, void* d_ws, size_t ws_size,
                              hipStream_t stream) {
  const float* x    = (const float*)d_in[0];
  const int*   mask = (const int*)d_in[1];
  const float* wq   = (const float*)d_in[2];
  const float* wk   = (const float*)d_in[3];
  const float* wv   = (const float*)d_in[4];
  const float* wo   = (const float*)d_in[5];
  const float* w1   = (const float*)d_in[6];
  const float* b1   = (const float*)d_in[7];
  const float* w2   = (const float*)d_in[8];
  const float* b2   = (const float*)d_in[9];
  const float* a1   = (const float*)d_in[10];
  const float* be1  = (const float*)d_in[11];
  const float* a2   = (const float*)d_in[12];
  const float* be2  = (const float*)d_in[13];
  float* dout = (float*)d_out;

  char* w = (char*)d_ws;
  auto take = [&](size_t bytes) -> char* {
    char* p = w;
    w += (bytes + 255) & ~(size_t)255;
    return p;
  };
  unsigned short* Wqkv_h = (unsigned short*)take((size_t)3072 * 1024 * 2);
  unsigned short* Wqkv_l = (unsigned short*)take((size_t)3072 * 1024 * 2);
  unsigned short* Wo_h   = (unsigned short*)take((size_t)1024 * 1024 * 2);
  unsigned short* Wo_l   = (unsigned short*)take((size_t)1024 * 1024 * 2);
  unsigned short* W1_h   = (unsigned short*)take((size_t)4096 * 1024 * 2);
  unsigned short* W1_l   = (unsigned short*)take((size_t)4096 * 1024 * 2);
  unsigned short* W2_h   = (unsigned short*)take((size_t)1024 * 4096 * 2);
  unsigned short* W2_l   = (unsigned short*)take((size_t)1024 * 4096 * 2);
  unsigned short* xn_h   = (unsigned short*)take((size_t)TT * DD * 2);
  unsigned short* xn_l   = (unsigned short*)take((size_t)TT * DD * 2);
  unsigned short* Qh     = (unsigned short*)take((size_t)BH * SS * DQ * 2);
  unsigned short* Kh     = (unsigned short*)take((size_t)BH * SS * DQ * 2);
  float*          Vf     = (float*)take((size_t)TT * DD * 4);
  float*          rm     = (float*)take((size_t)BH * SS * 4);
  float*          rl     = (float*)take((size_t)BH * SS * 4);
  unsigned short* at_h   = (unsigned short*)take((size_t)TT * DD * 2);
  unsigned short* at_l   = (unsigned short*)take((size_t)TT * DD * 2);
  float*          x1     = (float*)take((size_t)TT * DD * 4);
  unsigned short* h_h    = (unsigned short*)take((size_t)TT * 1024 * 2);
  unsigned short* h_l    = (unsigned short*)take((size_t)TT * 1024 * 2);

  // 1) weight conversion
  convert_wqkv_kernel<<<(3072 * 1024) / 256, 256, 0, stream>>>(wq, wk, wv, Wqkv_h, Wqkv_l);
  convert_direct_kernel<<<(1024 * 1024) / 256, 256, 0, stream>>>(wo, Wo_h, Wo_l, 1024 * 1024);
  convert_direct_kernel<<<(4096 * 1024) / 256, 256, 0, stream>>>(w1, W1_h, W1_l, 4096 * 1024);
  convert_direct_kernel<<<(1024 * 4096) / 256, 256, 0, stream>>>(w2, W2_h, W2_l, 1024 * 4096);

  // 2) LN1
  ln_split_kernel<<<TT, 256, 0, stream>>>(x, a1, be1, xn_h, xn_l);

  // 3) QKV projection (M=4096, N=3072, K=1024)
  gemm_bf16x3_kernel<M_QKV, DD, DD, DD><<<dim3(3072 / 128, TT / 128), 256, 0, stream>>>(
      xn_h, xn_l, Wqkv_h, Wqkv_l,
      Vf, nullptr, nullptr, nullptr, Qh, Kh);

  // 4) softmax row stats over full QK^T
  attn_stats_kernel<<<dim3(SS / 128, BH), 256, 0, stream>>>(Qh, Kh, mask, rm, rl);

  // 5) diagonal weight * V
  diag_v_kernel<<<(BH * SS) / 8, 256, 0, stream>>>(Qh, Kh, Vf, mask, rm, rl, at_h, at_l);

  // 6) output projection + residual; also seed d_out = x1 + b2
  gemm_bf16x3_kernel<M_ATTN, DD, DD, DD><<<dim3(1024 / 128, TT / 128), 256, 0, stream>>>(
      at_h, at_l, Wo_h, Wo_l,
      x1, dout, x, b2, nullptr, nullptr);

  // 7) LN2
  ln_split_kernel<<<TT, 256, 0, stream>>>(x1, a2, be2, xn_h, xn_l);

  // 8) FFN in 4 chunks of F=1024: h = relu(xn*W1c + b1c); d_out += h*W2c
  for (int c = 0; c < 4; c++) {
    gemm_bf16x3_kernel<M_H, DD, DD, DD><<<dim3(1024 / 128, TT / 128), 256, 0, stream>>>(
        xn_h, xn_l,
        W1_h + (size_t)c * 1024 * 1024, W1_l + (size_t)c * 1024 * 1024,
        nullptr, nullptr, nullptr, b1 + c * 1024, h_h, h_l);
    gemm_bf16x3_kernel<M_ACC, DD, FF, DD><<<dim3(1024 / 128, TT / 128), 256, 0, stream>>>(
        h_h, h_l,
        W2_h + (size_t)c * 1024, W2_l + (size_t)c * 1024,
        dout, nullptr, nullptr, nullptr, nullptr, nullptr);
  }
}
